// GRU_43044162241096
// MI455X (gfx1250) — matrix operand order repaired and checked by reading, not compiled, not run on hardware
//
#include <hip/hip_runtime.h>
#include <stdint.h>

// Problem constants (match reference)
#define B_    128
#define T_    1024
#define DIN   512
#define H_    1024
#define K1    1536            // DIN + H_ : combined K for [x | h] GEMMs
#define NWG   16              // workgroups per cluster (H column tiles of 64)
#define NCL   8               // clusters (batch tiles of 16 rows)
#define ROWS  16              // batch rows per cluster (WMMA M)
#define WGCOL 64              // H columns per workgroup
#define THREADS 128           // 4 waves of 32

typedef __bf16 bf16;
typedef bf16  v16bf __attribute__((ext_vector_type(16)));
typedef bf16  v8bf  __attribute__((ext_vector_type(8)));
typedef float v8f   __attribute__((ext_vector_type(8)));
typedef unsigned int  u32x4 __attribute__((ext_vector_type(4)));
typedef int           i32x8 __attribute__((ext_vector_type(8)));
typedef int           i32x4 __attribute__((ext_vector_type(4)));
typedef unsigned short ushort_t;

// ---- workspace layout (bytes) ------------------------------------------
#define WPK_OFF   0
#define WPK_BYTES ((size_t)3 * H_ * K1 * 2)          // 9,437,184
#define HBF_OFF   (WPK_OFF + WPK_BYTES)              // h state, bf16 [B_][H_]
#define RHB_OFF   (HBF_OFF + (size_t)B_ * H_ * 2)    // r*h,     bf16 [B_][H_]
#define CTR_OFF   (RHB_OFF + (size_t)B_ * H_ * 2)    // NCL spin counters, 256B stride

// ---- helpers ------------------------------------------------------------
static __device__ __forceinline__ ushort_t f2bf(float f) {
  unsigned u = __float_as_uint(f);
  u += 0x7FFFu + ((u >> 16) & 1u);   // round-to-nearest-even
  return (ushort_t)(u >> 16);
}

static __device__ __forceinline__ float sigmoidf_fast(float x) {
  x = fminf(fmaxf(x, -30.f), 30.f);
  return __builtin_amdgcn_rcpf(1.f + __expf(-x));     // v_rcp_f32, no IEEE div
}

static __device__ __forceinline__ float tanhf_fast(float x) {
  x = fminf(fmaxf(x, -15.f), 15.f);
  return 1.f - 2.f * __builtin_amdgcn_rcpf(__expf(2.f * x) + 1.f);
}

static __device__ __forceinline__ v8f wmma_bf16(v16bf a, v16bf b, v8f c) {
  return __builtin_amdgcn_wmma_f32_16x16x32_bf16(
      /*neg_a=*/false, a, /*neg_b=*/false, b,
      /*c_mod=*/(short)0, c, /*reuse_a=*/false, /*reuse_b=*/false);
}

// A fragment (16x32 bf16, MxK) from LDS row (per CDNA5 ISA 7.12.2 layout).
static __device__ __forceinline__ v16bf ldsA(const ushort_t* row, int kb, int half) {
  const v8bf* p0 = (const v8bf*)(row + kb + 8 * half);
  const v8bf* p1 = (const v8bf*)(row + kb + 16 + 8 * half);
  v8bf lo = *p0, hi = *p1;
  return __builtin_shufflevector(lo, hi, 0,1,2,3,4,5,6,7,8,9,10,11,12,13,14,15);
}

// B fragment (32x16 bf16, KxN) from global column-major weights.
static __device__ __forceinline__ v16bf gldB(const ushort_t* col, int kb, int half) {
  const v8bf* p0 = (const v8bf*)(col + kb + 8 * half);
  const v8bf* p1 = (const v8bf*)(col + kb + 16 + 8 * half);
  v8bf lo = *p0, hi = *p1;
  return __builtin_shufflevector(lo, hi, 0,1,2,3,4,5,6,7,8,9,10,11,12,13,14,15);
}

// Build TDM D# group0: count=1, lds_addr, 57-bit global addr, type=2 (image)
static __device__ __forceinline__ u32x4 tdm_g0(unsigned lds_addr, unsigned long long ga) {
  u32x4 g;
  g.x = 0x1u;                                          // count=1, user mode
  g.y = lds_addr;                                      // LDS byte offset
  g.z = (unsigned)(ga & 0xFFFFFFFFull);                // global_addr[31:0]
  g.w = (unsigned)((ga >> 32) & 0x01FFFFFFull) | 0x80000000u;  // [56:32] | type=2
  return g;
}

// TDM D# group1: 2D tile, data_size=4B, 512 dwords/row x 16 rows, stride 512
static __device__ __forceinline__ i32x8 tdm_g1(void) {
  i32x8 g;
  g[0] = (int)(2u << 16);       // workgroup_mask=0, data_size=2 (4 bytes)
  g[1] = (int)(512u << 16);     // tensor_dim0[15:0]=512 in bits[31:16]
  g[2] = (int)(16u << 16);      // tensor_dim0[31:16]=0 | tensor_dim1[15:0]=16
  g[3] = (int)(512u << 16);     // tensor_dim1[31:16]=0 | tile_dim0=512
  g[4] = 16;                    // tile_dim1=16, tile_dim2=0
  g[5] = 512;                   // tensor_dim0_stride[31:0]=512
  g[6] = 0;                     // stride[47:32]=0, tensor_dim1_stride lo=0
  g[7] = 0;
  return g;
}

// Intra-cluster barrier: CDNA5 cluster split-barrier (NOP when dispatched
// without clusters) + global-atomic arrive/spin fallback for plain dispatch.
static __device__ __forceinline__ void cluster_sync(unsigned* ctr, unsigned* epoch) {
  __threadfence();              // release
  __syncthreads();
  if (threadIdx.x == 0) {
    unsigned tgt = (++(*epoch)) * NWG;
    atomicAdd(ctr, 1u);
    while (atomicAdd(ctr, 0u) < tgt) __builtin_amdgcn_s_sleep(1);
  }
  __builtin_amdgcn_s_cluster_barrier();   // s_barrier_signal/-wait -3
  __syncthreads();
  __threadfence();              // acquire
}

// ---- kernel 1: pack weights f32 -> bf16, [W_g ; R_g] column-major -------
__global__ void gru_pack_weights(const float* __restrict__ Wz, const float* __restrict__ Wr,
                                 const float* __restrict__ Wh, const float* __restrict__ Rz,
                                 const float* __restrict__ Rr, const float* __restrict__ Rh,
                                 ushort_t* __restrict__ wpk) {
  size_t idx = (size_t)blockIdx.x * blockDim.x + threadIdx.x;   // (g,k,n), n innermost
  if (idx >= (size_t)3 * K1 * H_) return;
  int n = (int)(idx % H_);
  int k = (int)((idx / H_) % K1);
  int g = (int)(idx / ((size_t)H_ * K1));
  const float* W = (g == 0) ? Wz : (g == 1) ? Wr : Wh;
  const float* R = (g == 0) ? Rz : (g == 1) ? Rr : Rh;
  float v = (k < DIN) ? W[(size_t)k * H_ + n] : R[(size_t)(k - DIN) * H_ + n];
  wpk[((size_t)g * H_ + n) * K1 + k] = f2bf(v);
}

// ---- kernel 2: zero hidden state + barrier counters ---------------------
__global__ void gru_init(unsigned* __restrict__ hbf_u32, unsigned* __restrict__ ctrs) {
  size_t i = (size_t)blockIdx.x * blockDim.x + threadIdx.x;
  if (i < (size_t)B_ * H_ / 2) hbf_u32[i] = 0u;
  if (blockIdx.x == 0)
    for (unsigned j = threadIdx.x; j < NCL * 64; j += blockDim.x) ctrs[j] = 0u;
}

// ---- kernel 3: persistent GRU -------------------------------------------
__global__ __launch_bounds__(THREADS)
void gru_persistent(const float* __restrict__ x,
                    const float* __restrict__ Bz, const float* __restrict__ Br,
                    const float* __restrict__ Bh,
                    const ushort_t* __restrict__ wpk,
                    ushort_t* __restrict__ hbf, ushort_t* __restrict__ rhbf,
                    unsigned* __restrict__ ctrs,
                    float* __restrict__ out) {
  __shared__ ushort_t bufX[ROWS * DIN];  // 16 KB: x_t bf16, row stride DIN
  __shared__ ushort_t bufH[ROWS * H_];   // 32 KB: h (then r*h) bf16, TDM dest
  __shared__ unsigned epoch;

  const int tid  = threadIdx.x;
  const int wv   = tid >> 5;
  const int lane = tid & 31;
  const int half = lane >> 4;
  const int lr   = lane & 15;

  const int cid     = blockIdx.x >> 4;
  const int wgc     = blockIdx.x & 15;
  const int rowbase = cid * ROWS;
  const int gcol    = wgc * WGCOL + wv * 16 + lr;

  unsigned* ctr = ctrs + cid * 64;

  if (tid == 0) epoch = 0;
  __syncthreads();

  const ushort_t* pz = wpk + (size_t)(0 * H_ + gcol) * K1;
  const ushort_t* pr = wpk + (size_t)(1 * H_ + gcol) * K1;
  const ushort_t* ph = wpk + (size_t)(2 * H_ + gcol) * K1;
  const float bz = Bz[gcol], br = Br[gcol], bh = Bh[gcol];

  const ushort_t* arowX = &bufX[lr * DIN];
  const ushort_t* arowH = &bufH[lr * H_];

  // Loop-invariant TDM descriptors: 16x2048B tile global->LDS
  const unsigned ldsH = (unsigned)(uintptr_t)(&bufH[0]);
  const u32x4 g0h = tdm_g0(ldsH, (unsigned long long)(uintptr_t)(hbf  + (size_t)rowbase * H_));
  const u32x4 g0r = tdm_g0(ldsH, (unsigned long long)(uintptr_t)(rhbf + (size_t)rowbase * H_));
  const i32x8 g1  = tdm_g1();
  const i32x4 gz4 = {0, 0, 0, 0};
  const i32x8 gz8 = {0, 0, 0, 0, 0, 0, 0, 0};

  v8f hold = {};                         // h state, C/D layout: row = e + 8*half

  for (int t = 0; t < T_; ++t) {
    // TDM: h tile (global bf16 [16][1024]) -> bufH, wave 0 issues
    if (wv == 0) {
      __builtin_amdgcn_tensor_load_to_lds(g0h, g1, gz4, gz4, gz8, 0);
    }
    // stage x_t (f32 -> bf16) into bufX while TDM runs
    for (int i = tid; i < ROWS * (DIN / 4); i += THREADS) {
      int r = i >> 7;                    // 128 float4 chunks per row
      int c = (i & 127) * 4;
      float4 v = *(const float4*)(x + ((size_t)(rowbase + r) * T_ + t) * DIN + c);
      ushort4 o;
      o.x = f2bf(v.x); o.y = f2bf(v.y); o.z = f2bf(v.z); o.w = f2bf(v.w);
      *(ushort4*)(&bufX[r * DIN + c]) = o;
    }
    if (wv == 0) __builtin_amdgcn_s_wait_tensorcnt(0);
    __syncthreads();

    // phase 1: z,r = sigmoid(x@W + h@R + b)
    v8f az = {}, ar = {};
#pragma unroll 2
    for (int kb = 0; kb < DIN; kb += 32) {
      v16bf a = ldsA(arowX, kb, half);
      az = wmma_bf16(a, gldB(pz, kb, half), az);
      ar = wmma_bf16(a, gldB(pr, kb, half), ar);
    }
#pragma unroll 2
    for (int kb = 0; kb < H_; kb += 32) {
      v16bf a = ldsA(arowH, kb, half);
      az = wmma_bf16(a, gldB(pz, DIN + kb, half), az);
      ar = wmma_bf16(a, gldB(pr, DIN + kb, half), ar);
    }
    v8f zg, rg;
#pragma unroll
    for (int e = 0; e < 8; ++e) {
      zg[e] = sigmoidf_fast(az[e] + bz);
      rg[e] = sigmoidf_fast(ar[e] + br);
    }
    // publish r .* h_old (bf16) for the whole cluster
#pragma unroll
    for (int e = 0; e < 8; ++e) {
      int grow = rowbase + e + 8 * half;
      rhbf[(size_t)grow * H_ + gcol] = f2bf(rg[e] * hold[e]);
    }
    cluster_sync(ctr, &epoch);

    // TDM: r*h tile -> bufH (phase-1 reads of bufH finished before sync)
    if (wv == 0) {
      __builtin_amdgcn_tensor_load_to_lds(g0r, g1, gz4, gz4, gz8, 0);
      __builtin_amdgcn_s_wait_tensorcnt(0);
    }
    __syncthreads();

    // phase 2: h~ = tanh(x@W_h + (r*h)@R_h + b_h); h = (1-z)h + z h~
    v8f ah = {};
#pragma unroll 2
    for (int kb = 0; kb < DIN; kb += 32) {
      v16bf a = ldsA(arowX, kb, half);
      ah = wmma_bf16(a, gldB(ph, kb, half), ah);
    }
#pragma unroll 2
    for (int kb = 0; kb < H_; kb += 32) {
      v16bf a = ldsA(arowH, kb, half);
      ah = wmma_bf16(a, gldB(ph, DIN + kb, half), ah);
    }
#pragma unroll
    for (int e = 0; e < 8; ++e) {
      float ht = tanhf_fast(ah[e] + bh);
      float hn = (1.f - zg[e]) * hold[e] + zg[e] * ht;
      hold[e] = hn;
      int grow = rowbase + e + 8 * half;
      hbf[(size_t)grow * H_ + gcol] = f2bf(hn);
      if (t == T_ - 1) out[(size_t)grow * H_ + gcol] = hn;
    }
    cluster_sync(ctr, &epoch);
  }
}

// ---- host-side launcher --------------------------------------------------
extern "C" void kernel_launch(void* const* d_in, const int* in_sizes, int n_in,
                              void* d_out, int out_size, void* d_ws, size_t ws_size,
                              hipStream_t stream) {
  (void)in_sizes; (void)n_in; (void)out_size; (void)ws_size;
  const float* x  = (const float*)d_in[0];
  const float* Wz = (const float*)d_in[1];
  const float* Wr = (const float*)d_in[2];
  const float* Wh = (const float*)d_in[3];
  const float* Rz = (const float*)d_in[4];
  const float* Rr = (const float*)d_in[5];
  const float* Rh = (const float*)d_in[6];
  const float* Bz = (const float*)d_in[7];
  const float* Br = (const float*)d_in[8];
  const float* Bh = (const float*)d_in[9];

  char* ws = (char*)d_ws;
  ushort_t* wpk  = (ushort_t*)(ws + WPK_OFF);
  ushort_t* hbf  = (ushort_t*)(ws + HBF_OFF);
  ushort_t* rhb  = (ushort_t*)(ws + RHB_OFF);
  unsigned* ctrs = (unsigned*)(ws + CTR_OFF);

  {
    size_t total = (size_t)3 * K1 * H_;
    int grid = (int)((total + 255) / 256);
    gru_pack_weights<<<grid, 256, 0, stream>>>(Wz, Wr, Wh, Rz, Rr, Rh, wpk);
  }
  gru_init<<<(B_ * H_ / 2 + 255) / 256, 256, 0, stream>>>((unsigned*)hbf, ctrs);
  gru_persistent<<<NCL * NWG, THREADS, 0, stream>>>(
      x, Bz, Br, Bh, wpk, hbf, rhb, ctrs, (float*)d_out);
}
